// LSTM_33303176413424
// MI455X (gfx1250) — compile-verified, matrix operand order checked by reading, not run on hardware
//
#include <hip/hip_runtime.h>
#include <hip/hip_bf16.h>

// ---------------------------------------------------------------------------
// 2-layer LSTM for MI455X (gfx1250, wave32, WMMA).
//  - bf16 weights (converted per call into d_ws), f32 accumulate via
//    v_wmma_f32_16x16x32_bf16.
//  - One kernel launch per time step; grid.y=2 pipelines layer0@t with
//    layer1@(t-1).
//  - Each workgroup covers TWO 16-row batch tiles (32 rows) so every B
//    (weight) fragment loaded from L2 feeds two WMMAs: weight traffic per
//    step is halved vs one-Mtile-per-WG (weights are read 2x, not 4x).
//  - Each wave: 2 Mtiles x 4 gates = 8 f32 accumulators; LSTM cell update
//    (sigmoid/tanh via v_exp2/v_rcp) done fully in-register.
// Workspace (~35.1 MB):
//   wih_bf16[2][4096][1024], whh_bf16[2][4096][1024],
//   h0 pingpong[2][64][1024] f32, h1 pingpong[2][64][1024] f32,
//   c0[64][1024] f32, c1[64][1024] f32
// ---------------------------------------------------------------------------

#define INPUTS  1024
#define HIDDEN  1024
#define BATCH   64
#define SEQLEN  512
#define GATES4  4096            // 4*HIDDEN
#define LDSW    2056            // 2048 + 8 pad (u16) -> breaks 64-bank alignment

typedef __attribute__((ext_vector_type(16))) __bf16         v16bf;
typedef __attribute__((ext_vector_type(8)))  float          v8f;
typedef __attribute__((ext_vector_type(4)))  float          f32x4;
typedef __attribute__((ext_vector_type(4)))  unsigned int   u32x4;
typedef __attribute__((ext_vector_type(4)))  unsigned short u16x4;

union Frag16 { v16bf v; u32x4 q[2]; };

__device__ __forceinline__ unsigned short f2bf(float f) {
    union { float f; unsigned u; } v; v.f = f;
    unsigned r = (v.u + 0x7FFFu + ((v.u >> 16) & 1u)) >> 16;
    return (unsigned short)r;
}

__device__ __forceinline__ float fast_sigmoid(float x) {
    float e = __builtin_amdgcn_exp2f(-1.4426950408889634f * x);
    return __builtin_amdgcn_rcpf(1.0f + e);
}
__device__ __forceinline__ float fast_tanh(float x) {
    float e = __builtin_amdgcn_exp2f(2.8853901817779268f * x);
    return 1.0f - 2.0f * __builtin_amdgcn_rcpf(e + 1.0f);
}

// ---------------------------------------------------------------------------
__global__ void lstm_cvt_bf16(const float* __restrict__ src,
                              unsigned short* __restrict__ dst, int n) {
    int i = blockIdx.x * 256 + threadIdx.x;
    if (i < n) dst[i] = f2bf(src[i]);
}

__global__ void lstm_zero_f32(float* __restrict__ p, int n) {
    int i = blockIdx.x * 256 + threadIdx.x;
    if (i < n) p[i] = 0.0f;
}

// ---------------------------------------------------------------------------
// blockIdx.y: 0 -> layer0 @ t=k (skip k==SEQLEN); 1 -> layer1 @ t=k-1 (skip k==0)
// blockIdx.x in [0,16): mpair = bx>>3 (32 batch rows), hgroup = bx&7.
// 256 threads = 8 waves; wave w owns hidden tile (hgroup*8 + w), both Mtiles,
// all four gates.
__global__ void __launch_bounds__(256)
lstm_step(const float* __restrict__ x,
          const float* __restrict__ bih,
          const float* __restrict__ bhh,
          float* __restrict__ out,
          const unsigned short* __restrict__ wih,
          const unsigned short* __restrict__ whh,
          float* __restrict__ h0, float* __restrict__ h1,
          float* __restrict__ c0, float* __restrict__ c1,
          int k)
{
    const int layer = blockIdx.y;
    int t;
    if (layer == 0) { if (k >= SEQLEN) return; t = k; }
    else            { if (k == 0)      return; t = k - 1; }

    const int mpair  = blockIdx.x >> 3;
    const int hgroup = blockIdx.x & 7;
    const int mbase  = mpair * 32;          // 32 batch rows per workgroup

    float* hstate = layer ? h1 : h0;
    const float* hprev = hstate + (size_t)(t & 1)       * (BATCH * HIDDEN);
    float*       hout  = hstate + (size_t)((t + 1) & 1) * (BATCH * HIDDEN);
    float*       cbuf  = layer ? c1 : c0;
    // layer1 input = h0_t, written by slot0 last launch with parity (t+1)&1
    const float* l1in  = h0 + (size_t)((t + 1) & 1) * (BATCH * HIDDEN);

    __shared__ unsigned short smem[32 * LDSW];

    // ---- stage A panel: 32 batch rows; cols 0..1023 = x_t, 1024..2047 = h_{t-1}
    const int tid = threadIdx.x;
    #pragma unroll 8
    for (int i = 0; i < 64; ++i) {
        int gidx = tid + (i << 8);          // 0..16383 float4-groups
        int r    = gidx >> 9;               // row 0..31
        int col  = (gidx & 511) << 2;       // col 0..2044, step 4
        int b    = mbase + r;
        const float* src; int sc;
        if (col < INPUTS) {
            src = (layer == 0) ? (x + ((size_t)b * SEQLEN + t) * INPUTS)
                               : (l1in + (size_t)b * HIDDEN);
            sc = col;
        } else {
            src = hprev + (size_t)b * HIDDEN;
            sc  = col - INPUTS;
        }
        f32x4 v = *(const f32x4*)(src + sc);
        u16x4 pk;
        pk[0] = f2bf(v[0]); pk[1] = f2bf(v[1]);
        pk[2] = f2bf(v[2]); pk[3] = f2bf(v[3]);
        *(u16x4*)&smem[r * LDSW + col] = pk;
    }
    __syncthreads();

    // ---- per-wave WMMA: 2 Mtiles x 4 gates, K = 2048 (1024 ih + 1024 hh)
    const int wave   = tid >> 5;
    const int lane   = tid & 31;
    const int lane15 = lane & 15;
    const int hi8    = (lane >> 4) & 1;     // lanes 16-31 take the K+8 halves
    const int htile  = hgroup * 8 + wave;
    const int hcol0  = htile * 16;

    const unsigned short* wihL = wih + (size_t)layer * GATES4 * HIDDEN;
    const unsigned short* whhL = whh + (size_t)layer * GATES4 * HIDDEN;

    const unsigned short* browI[4];
    const unsigned short* browH[4];
    #pragma unroll
    for (int g = 0; g < 4; ++g) {
        int wr = g * HIDDEN + hcol0 + lane15;
        browI[g] = wihL + (size_t)wr * HIDDEN;
        browH[g] = whhL + (size_t)wr * HIDDEN;
        __builtin_prefetch(browI[g], 0, 0);
        __builtin_prefetch(browH[g], 0, 0);
    }

    v8f acc[2][4];
    #pragma unroll
    for (int m = 0; m < 2; ++m)
        #pragma unroll
        for (int g = 0; g < 4; ++g)
            acc[m][g] = (v8f){0.f,0.f,0.f,0.f,0.f,0.f,0.f,0.f};

    const unsigned short* arow0 = &smem[(lane15)      * LDSW];
    const unsigned short* arow1 = &smem[(16 + lane15) * LDSW];

    #pragma unroll
    for (int half = 0; half < 2; ++half) {
        const unsigned short* const* brow = half ? browH : browI;
        for (int kk = 0; kk < 32; ++kk) {
            const int ac = half * 1024 + kk * 32 + hi8 * 8;  // LDS column base
            Frag16 a0, a1;
            a0.q[0] = *(const u32x4*)(arow0 + ac);
            a0.q[1] = *(const u32x4*)(arow0 + ac + 16);
            a1.q[0] = *(const u32x4*)(arow1 + ac);
            a1.q[1] = *(const u32x4*)(arow1 + ac + 16);

            const int kb = kk * 32 + hi8 * 8;
            #pragma unroll
            for (int g = 0; g < 4; ++g) {
                const unsigned short* bp = brow[g] + kb;
                Frag16 bf;
                bf.q[0] = *(const u32x4*)bp;
                bf.q[1] = *(const u32x4*)(bp + 16);
                acc[0][g] = __builtin_amdgcn_wmma_f32_16x16x32_bf16(
                                false, a0.v, false, bf.v, (short)0, acc[0][g],
                                false, false);
                acc[1][g] = __builtin_amdgcn_wmma_f32_16x16x32_bf16(
                                false, a1.v, false, bf.v, (short)0, acc[1][g],
                                false, false);
            }
        }
    }

    // ---- LSTM cell update in-register.
    // C/D layout: lane -> col = hcol0 + lane15; VGPR r -> row (hi8*8 + r)
    const int col = hcol0 + lane15;
    float bias[4];
    #pragma unroll
    for (int g = 0; g < 4; ++g)
        bias[g] = bih[layer * GATES4 + g * HIDDEN + col] +
                  bhh[layer * GATES4 + g * HIDDEN + col];

    const size_t BSH = (size_t)BATCH * SEQLEN * HIDDEN;
    #pragma unroll
    for (int m = 0; m < 2; ++m) {
        #pragma unroll
        for (int r = 0; r < 8; ++r) {
            int b = mbase + m * 16 + hi8 * 8 + r;
            size_t ci = (size_t)b * HIDDEN + col;
            float i_g = fast_sigmoid(acc[m][0][r] + bias[0]);
            float f_g = fast_sigmoid(acc[m][1][r] + bias[1]);
            float g_g = fast_tanh   (acc[m][2][r] + bias[2]);
            float o_g = fast_sigmoid(acc[m][3][r] + bias[3]);
            float cold = cbuf[ci];
            float cn = f_g * cold + i_g * g_g;
            float hn = o_g * fast_tanh(cn);
            cbuf[ci] = cn;
            hout[ci] = hn;
            if (layer == 1) {
                out[((size_t)b * SEQLEN + t) * HIDDEN + col] = hn;
                if (t == SEQLEN - 1) {
                    out[BSH + ci] = hn;                           // h_last
                    out[BSH + (size_t)BATCH * HIDDEN + ci] = cn;  // c_last
                }
            }
        }
    }
}

// ---------------------------------------------------------------------------
extern "C" void kernel_launch(void* const* d_in, const int* in_sizes, int n_in,
                              void* d_out, int out_size, void* d_ws, size_t ws_size,
                              hipStream_t stream)
{
    const float* x    = (const float*)d_in[0];
    const float* W_ih = (const float*)d_in[1];
    const float* b_ih = (const float*)d_in[2];
    const float* W_hh = (const float*)d_in[3];
    const float* b_hh = (const float*)d_in[4];
    float* out = (float*)d_out;

    const int WN = 2 * GATES4 * HIDDEN;                 // elems per weight set
    unsigned short* wih = (unsigned short*)d_ws;
    unsigned short* whh = wih + (size_t)WN;
    float* h0 = (float*)(whh + (size_t)WN);
    float* h1 = h0 + 2 * BATCH * HIDDEN;
    float* c0 = h1 + 2 * BATCH * HIDDEN;
    float* c1 = c0 + BATCH * HIDDEN;

    lstm_cvt_bf16<<<(WN + 255) / 256, 256, 0, stream>>>(W_ih, wih, WN);
    lstm_cvt_bf16<<<(WN + 255) / 256, 256, 0, stream>>>(W_hh, whh, WN);

    const int ZN = 6 * BATCH * HIDDEN;
    lstm_zero_f32<<<(ZN + 255) / 256, 256, 0, stream>>>(h0, ZN);

    // pipelined time loop: launch k does layer0@t=k and layer1@t=k-1
    for (int k = 0; k <= SEQLEN; ++k) {
        lstm_step<<<dim3(16, 2, 1), 256, 0, stream>>>(
            x, b_ih, b_hh, out, wih, whh, h0, h1, c0, c1, k);
    }
}
